// TestModel_21139829031169
// MI455X (gfx1250) — matrix operand order repaired and checked by reading, not compiled, not run on hardware
//
#include <hip/hip_runtime.h>
#include <math.h>

// ---------------- types ----------------
typedef __attribute__((ext_vector_type(16))) _Float16 v16h;
typedef __attribute__((ext_vector_type(8)))  _Float16 v8h;
typedef __attribute__((ext_vector_type(8)))  float    v8f;

#define NN 24000           // N_NODE
#define NGRAPH 8
#define EPSF 1e-5f
#define SQRT2F 1.41421356237f
#define SQRT3F 1.73205080757f
#define INV_SQRT20 0.22360679775f   // weight scale 1/sqrt(20)
#define C1F 0.44721359550f          // (1/5)^0.5
#define C2F 0.22360679775f          // (1/20)^0.5
#define C2S3F 0.12909944487f        // c2/sqrt(3)

// workspace layout (floats)
#define OFF_F    0                   // NN*40
#define OFF_G    (NN*40)             // NN*20
#define OFF_S1   (NN*60)             // 30 (pad to 32)
#define OFF_S2   (NN*60 + 32)        // 40
#define OFF_OACC (NN*60 + 72)        // 160
#define WS_FLOATS (NN*60 + 72 + 160)

// ---------------- helpers ----------------
__device__ __forceinline__ void atomAddF(float* p, float v) {
  __hip_atomic_fetch_add(p, v, __ATOMIC_RELAXED, __HIP_MEMORY_SCOPE_AGENT);
}

__device__ __forceinline__ v8f wmma_f16(v16h a, v16h b, v8f c) {
  // D = A(16x32 f16) x B(32x16 f16) + C(16x16 f32)
  return __builtin_amdgcn_wmma_f32_16x16x32_f16(
      /*neg_a=*/false, a, /*neg_b=*/false, b,
      /*c_mod=*/(short)0, c, /*reuse_a=*/false, /*reuse_b=*/false);
}

// A matrix 16x32 f16 fragment per ISA layout:
// lane<16: M=lane, K halves = {b..b+7, 16+b..16+b+7}, b = 0
// lane>=16: same M, b = 8
__device__ __forceinline__ v16h load_A_frag(const _Float16* At, int lane) {
  int m = lane & 15;
  int b = (lane >= 16) ? 8 : 0;
  union { v16h v; v8h h[2]; } u;
  u.h[0] = *(const v8h*)(At + m * 32 + b);
  u.h[1] = *(const v8h*)(At + m * 32 + 16 + b);
  return u.v;
}

// B matrix 32x16 f16 fragment; B stored in LDS as [N][32] (K contiguous)
// lane<16: N=lane, K = {0..7, 16..23} ; lane>=16: N=lane-16, K = {8..15, 24..31}
__device__ __forceinline__ v16h load_B_frag(const _Float16* Bt, int tile, int lane) {
  int n = lane & 15;
  int b = (lane >= 16) ? 8 : 0;
  const _Float16* row = Bt + (tile * 16 + n) * 32;
  union { v16h v; v8h h[2]; } u;
  u.h[0] = *(const v8h*)(row + b);
  u.h[1] = *(const v8h*)(row + 16 + b);
  return u.v;
}

// stage f32 weights [K=20][Nreal] (pre-scaled by 1/sqrt(20)) into LDS f16 [Npad][32]
__device__ __forceinline__ void stage_weights(const float* Wg, int Nreal, int Npad,
                                              _Float16* Bl, int tid, int nth) {
  for (int idx = tid; idx < Npad * 32; idx += nth) {
    int n = idx >> 5, k = idx & 31;
    float v = (k < 20 && n < Nreal) ? Wg[k * Nreal + n] * INV_SQRT20 : 0.0f;
    Bl[idx] = (_Float16)v;
  }
}

// per-edge geometry + radial basis embedding (writes one 32-wide A row)
__device__ __forceinline__ void edge_prep(const float* pos, int src, int dst, bool valid,
                                          _Float16* Arow, float sh1[3]) {
  float dx = pos[dst * 3 + 0] - pos[src * 3 + 0];
  float dy = pos[dst * 3 + 1] - pos[src * 3 + 1];
  float dz = pos[dst * 3 + 2] - pos[src * 3 + 2];
  float elen = sqrtf(dx * dx + dy * dy + dz * dz);
  float inv = 1.0f / elen;
  sh1[0] = SQRT3F * dy * inv;   // unit[[1,2,0]]
  sh1[1] = SQRT3F * dz * inv;
  sh1[2] = SQRT3F * dx * inv;
  const float CEMB = 1.14136f * 7.38905609893f * 4.47213595500f; // 1.14136*e^2*sqrt(20)
  float t = elen * 52.5f;       // elen / step, step = R/21, R=0.4
  #pragma unroll
  for (int i = 0; i < 20; i++) {
    float diff = t - (float)(i + 1);
    float a = diff + 1.0f;
    float b = 1.0f - diff;
    float s1 = (a > 0.0f) ? __expf(-1.0f / a) : 0.0f;
    float s2 = (b > 0.0f) ? __expf(-1.0f / b) : 0.0f;
    Arow[i] = (_Float16)(valid ? (CEMB * s1 * s2) : 0.0f);
  }
  #pragma unroll
  for (int i = 20; i < 32; i++) Arow[i] = (_Float16)0.0f;
}

// ---------------- kernels ----------------
__global__ void zero_kernel(float* p, int n) {
  for (int i = blockIdx.x * blockDim.x + threadIdx.x; i < n; i += gridDim.x * blockDim.x)
    p[i] = 0.0f;
}

#define W1WAVES 6
__global__ void __launch_bounds__(32 * W1WAVES)
pass1_kernel(const float* __restrict__ pos, const float* __restrict__ xel,
             const float* __restrict__ w1, const float* __restrict__ w2,
             const int* __restrict__ esrc, const int* __restrict__ edst,
             float* __restrict__ fbuf, int E) {
  extern __shared__ char smem[];
  _Float16* B1 = (_Float16*)smem;                 // 32*32
  _Float16* B2 = B1 + 32 * 32;                    // 112*32
  _Float16* Atiles = B2 + 112 * 32;               // W1WAVES * 16*32
  float* wtiles = (float*)(Atiles + W1WAVES * 16 * 32); // W1WAVES * 16*112 (f32)

  int tid = threadIdx.x;
  stage_weights(w1, 20, 32, B1, tid, 32 * W1WAVES);
  stage_weights(w2, 100, 112, B2, tid, 32 * W1WAVES);
  __syncthreads();

  int lane = tid & 31;
  int wave = tid >> 5;
  int tile = blockIdx.x * W1WAVES + wave;
  if (tile * 16 >= E) return;                     // wave-uniform exit
  _Float16* At = Atiles + wave * 16 * 32;
  float* wt = wtiles + wave * 16 * 112;

  int e = tile * 16 + (lane & 15);
  bool valid = (e < E);
  int ec = valid ? e : (E - 1);
  int src = esrc[ec];
  int dst = edst[ec];
  float sh1[3] = {0.f, 0.f, 0.f};
  float x[5] = {0.f, 0.f, 0.f, 0.f, 0.f};
  if (lane < 16) {
    edge_prep(pos, src, dst, valid, At + (lane & 15) * 32, sh1);
    #pragma unroll
    for (int u = 0; u < 5; ++u) x[u] = xel[src * 5 + u];
  }
  asm volatile("s_wait_dscnt 0" ::: "memory");

  v8f cz = {};
  v16h a = load_A_frag(At, lane);
  v8f c0 = wmma_f16(a, load_B_frag(B1, 0, lane), cz);
  v8f c1 = wmma_f16(a, load_B_frag(B1, 1, lane), cz);

  // relu * sqrt2, restage hidden as next A matrix (16x32, cols 20..31 are zero)
  int moff = (lane >= 16) ? 8 : 0;
  int ncol = lane & 15;
  #pragma unroll
  for (int r = 0; r < 8; r++) {
    At[(r + moff) * 32 + ncol]      = (_Float16)(fmaxf(c0[r], 0.0f) * SQRT2F);
    At[(r + moff) * 32 + 16 + ncol] = (_Float16)(fmaxf(c1[r], 0.0f) * SQRT2F);
  }
  asm volatile("s_wait_dscnt 0" ::: "memory");
  v16h a2 = load_A_frag(At, lane);

  #pragma unroll
  for (int t = 0; t < 7; t++) {
    v8f c = wmma_f16(a2, load_B_frag(B2, t, lane), cz);
    #pragma unroll
    for (int r = 0; r < 8; r++)
      wt[(r + moff) * 112 + 16 * t + ncol] = c[r];
  }
  asm volatile("s_wait_dscnt 0" ::: "memory");

  if (lane < 16 && valid) {
    const float* wr = wt + (lane & 15) * 112;
    float* fd = fbuf + dst * 40;
    #pragma unroll
    for (int j = 0; j < 10; j++) {
      float o0 = 0.0f, t1 = 0.0f;
      #pragma unroll
      for (int u = 0; u < 5; u++) {
        o0 += wr[u * 10 + j] * x[u];
        t1 += wr[50 + u * 10 + j] * x[u];
      }
      o0 *= C1F; t1 *= C1F;
      atomAddF(fd + j, o0);
      atomAddF(fd + 10 + j * 3 + 0, t1 * sh1[0]);
      atomAddF(fd + 10 + j * 3 + 1, t1 * sh1[1]);
      atomAddF(fd + 10 + j * 3 + 2, t1 * sh1[2]);
    }
  }
}

__global__ void bn1_stats_kernel(const float* __restrict__ f, float* __restrict__ stats) {
  float acc[30];
  #pragma unroll
  for (int c = 0; c < 30; c++) acc[c] = 0.0f;
  for (int i = blockIdx.x * blockDim.x + threadIdx.x; i < NN; i += gridDim.x * blockDim.x) {
    const float* r = f + i * 40;
    #pragma unroll
    for (int c = 0; c < 10; c++) { float v = r[c]; acc[c] += v; acc[10 + c] += v * v; }
    #pragma unroll
    for (int c = 0; c < 10; c++) {
      #pragma unroll
      for (int d = 0; d < 3; d++) { float v = r[10 + c * 3 + d]; acc[20 + c] += v * v; }
    }
  }
  #pragma unroll
  for (int c = 0; c < 30; c++) {
    float v = acc[c];
    for (int o = 16; o > 0; o >>= 1) v += __shfl_down(v, o, 32);
    if ((threadIdx.x & 31) == 0) atomAddF(&stats[c], v);
  }
}

__global__ void bn1_apply_kernel(float* __restrict__ f, const float* __restrict__ stats) {
  int i = blockIdx.x * blockDim.x + threadIdx.x;
  if (i >= NN) return;
  float* r = f + i * 40;
  #pragma unroll
  for (int c = 0; c < 10; c++) {
    float mean = stats[c] * (1.0f / NN);
    float var = stats[10 + c] * (1.0f / NN) - mean * mean;
    r[c] = (r[c] - mean) * rsqrtf(var + EPSF);
  }
  #pragma unroll
  for (int c = 0; c < 10; c++) {
    float vs = rsqrtf(stats[20 + c] * (1.0f / (3.0f * NN)) + EPSF);
    #pragma unroll
    for (int d = 0; d < 3; d++) r[10 + c * 3 + d] *= vs;
  }
}

#define W2WAVES 2
__global__ void __launch_bounds__(32 * W2WAVES)
pass2_kernel(const float* __restrict__ pos, const float* __restrict__ fbuf,
             const float* __restrict__ w1, const float* __restrict__ w2,
             const int* __restrict__ esrc, const int* __restrict__ edst,
             float* __restrict__ gbuf, int E) {
  extern __shared__ char smem[];
  _Float16* B1 = (_Float16*)smem;                 // 32*32
  _Float16* B2 = B1 + 32 * 32;                    // 400*32
  _Float16* Atiles = B2 + 400 * 32;               // W2WAVES * 16*32
  _Float16* wtiles = Atiles + W2WAVES * 16 * 32;  // W2WAVES * 16*400 (f16)

  int tid = threadIdx.x;
  stage_weights(w1, 20, 32, B1, tid, 32 * W2WAVES);
  stage_weights(w2, 400, 400, B2, tid, 32 * W2WAVES);
  __syncthreads();

  int lane = tid & 31;
  int wave = tid >> 5;
  int tile = blockIdx.x * W2WAVES + wave;
  if (tile * 16 >= E) return;
  _Float16* At = Atiles + wave * 16 * 32;
  _Float16* wt = wtiles + wave * 16 * 400;

  int e = tile * 16 + (lane & 15);
  bool valid = (e < E);
  int ec = valid ? e : (E - 1);
  int src = esrc[ec];
  int dst = edst[ec];
  float sh1[3] = {0.f, 0.f, 0.f};
  float s[10], dvec[10];
  #pragma unroll
  for (int u = 0; u < 10; u++) { s[u] = 0.f; dvec[u] = 0.f; }
  if (lane < 16) {
    edge_prep(pos, src, dst, valid, At + (lane & 15) * 32, sh1);
    const float* fr = fbuf + src * 40;
    #pragma unroll
    for (int u = 0; u < 10; u++) {
      s[u] = fr[u];
      dvec[u] = fr[10 + u * 3 + 0] * sh1[0] +
                fr[10 + u * 3 + 1] * sh1[1] +
                fr[10 + u * 3 + 2] * sh1[2];
    }
  }
  asm volatile("s_wait_dscnt 0" ::: "memory");

  v8f cz = {};
  v16h a = load_A_frag(At, lane);
  v8f c0 = wmma_f16(a, load_B_frag(B1, 0, lane), cz);
  v8f c1 = wmma_f16(a, load_B_frag(B1, 1, lane), cz);

  int moff = (lane >= 16) ? 8 : 0;
  int ncol = lane & 15;
  #pragma unroll
  for (int r = 0; r < 8; r++) {
    At[(r + moff) * 32 + ncol]      = (_Float16)(fmaxf(c0[r], 0.0f) * SQRT2F);
    At[(r + moff) * 32 + 16 + ncol] = (_Float16)(fmaxf(c1[r], 0.0f) * SQRT2F);
  }
  asm volatile("s_wait_dscnt 0" ::: "memory");
  v16h a2 = load_A_frag(At, lane);

  #pragma unroll
  for (int t = 0; t < 25; t++) {
    v8f c = wmma_f16(a2, load_B_frag(B2, t, lane), cz);
    #pragma unroll
    for (int r = 0; r < 8; r++)
      wt[(r + moff) * 400 + 16 * t + ncol] = (_Float16)c[r];
  }
  asm volatile("s_wait_dscnt 0" ::: "memory");

  if (lane < 16 && valid) {
    const _Float16* wr = wt + (lane & 15) * 400;
    float* gd = gbuf + dst * 20;
    #pragma unroll
    for (int j = 0; j < 20; j++) {
      float o = 0.0f, o2 = 0.0f;
      #pragma unroll
      for (int u = 0; u < 10; u++) {
        o  += (float)wr[u * 20 + j] * s[u];
        o2 += (float)wr[200 + u * 20 + j] * dvec[u];
      }
      atomAddF(gd + j, C2F * o + C2S3F * o2);
    }
  }
}

__global__ void bn2_stats_kernel(const float* __restrict__ g, float* __restrict__ stats) {
  float acc[40];
  #pragma unroll
  for (int c = 0; c < 40; c++) acc[c] = 0.0f;
  for (int i = blockIdx.x * blockDim.x + threadIdx.x; i < NN; i += gridDim.x * blockDim.x) {
    const float* r = g + i * 20;
    #pragma unroll
    for (int c = 0; c < 20; c++) { float v = r[c]; acc[c] += v; acc[20 + c] += v * v; }
  }
  #pragma unroll
  for (int c = 0; c < 40; c++) {
    float v = acc[c];
    for (int o = 16; o > 0; o >>= 1) v += __shfl_down(v, o, 32);
    if ((threadIdx.x & 31) == 0) atomAddF(&stats[c], v);
  }
}

__global__ void finalize_kernel(const float* __restrict__ g, const float* __restrict__ stats,
                                const int* __restrict__ batch, float* __restrict__ oacc) {
  int i = blockIdx.x * blockDim.x + threadIdx.x;
  if (i >= NN) return;
  int b = batch[i];
  const float* gr = g + i * 20;
  #pragma unroll
  for (int c = 0; c < 20; c++) {
    float mean = stats[c] * (1.0f / NN);
    float var = stats[20 + c] * (1.0f / NN) - mean * mean;
    float gn = (gr[c] - mean) * rsqrtf(var + EPSF);
    atomAddF(oacc + b * 20 + c, gn);
  }
}

__global__ void softmax_kernel(const float* __restrict__ oacc, float* __restrict__ out) {
  int gidx = threadIdx.x;
  if (gidx >= NGRAPH) return;
  const float* r = oacc + gidx * 20;
  float m = r[0];
  for (int c = 1; c < 20; c++) m = fmaxf(m, r[c]);
  float ex[20];
  float sum = 0.0f;
  for (int c = 0; c < 20; c++) { ex[c] = __expf(r[c] - m); sum += ex[c]; }
  float inv = 1.0f / sum;
  for (int c = 0; c < 20; c++) out[gidx * 20 + c] = ex[c] * inv;
}

// ---------------- host ----------------
extern "C" void kernel_launch(void* const* d_in, const int* in_sizes, int n_in,
                              void* d_out, int out_size, void* d_ws, size_t ws_size,
                              hipStream_t stream) {
  const float* pos   = (const float*)d_in[0];
  const float* xel   = (const float*)d_in[1];
  const float* m1w1  = (const float*)d_in[2];
  const float* m1w2  = (const float*)d_in[3];
  const float* m2w1  = (const float*)d_in[4];
  const float* m2w2  = (const float*)d_in[5];
  const int*   batch = (const int*)d_in[6];
  const int*   esrc  = (const int*)d_in[7];
  const int*   edst  = (const int*)d_in[8];
  int E = in_sizes[7];

  float* ws   = (float*)d_ws;
  float* fbuf = ws + OFF_F;
  float* gbuf = ws + OFF_G;
  float* s1   = ws + OFF_S1;
  float* s2   = ws + OFF_S2;
  float* oacc = ws + OFF_OACC;
  float* out  = (float*)d_out;

  zero_kernel<<<256, 256, 0, stream>>>(ws, WS_FLOATS);

  int tiles = (E + 15) / 16;
  if (tiles > 0) {
    size_t sm1 = (size_t)(32 * 32 + 112 * 32 + W1WAVES * 16 * 32) * sizeof(_Float16)
               + (size_t)(W1WAVES * 16 * 112) * sizeof(float);
    pass1_kernel<<<(tiles + W1WAVES - 1) / W1WAVES, 32 * W1WAVES, sm1, stream>>>(
        pos, xel, m1w1, m1w2, esrc, edst, fbuf, E);
  }
  bn1_stats_kernel<<<94, 256, 0, stream>>>(fbuf, s1);
  bn1_apply_kernel<<<(NN + 255) / 256, 256, 0, stream>>>(fbuf, s1);
  if (tiles > 0) {
    size_t sm2 = (size_t)(32 * 32 + 400 * 32 + W2WAVES * 16 * 32 + W2WAVES * 16 * 400)
               * sizeof(_Float16);
    pass2_kernel<<<(tiles + W2WAVES - 1) / W2WAVES, 32 * W2WAVES, sm2, stream>>>(
        pos, fbuf, m2w1, m2w2, esrc, edst, gbuf, E);
  }
  bn2_stats_kernel<<<94, 256, 0, stream>>>(gbuf, s2);
  finalize_kernel<<<(NN + 255) / 256, 256, 0, stream>>>(gbuf, s2, batch, oacc);
  softmax_kernel<<<1, 32, 0, stream>>>(oacc, out);
  (void)n_in; (void)out_size; (void)ws_size;
}
